// DE_Conv_56427280335471
// MI455X (gfx1250) — matrix operand order repaired and checked by reading, not compiled, not run on hardware
//
#include <hip/hip_runtime.h>

typedef __attribute__((ext_vector_type(16))) _Float16 v16h;
typedef __attribute__((ext_vector_type(8)))  float    v8f;

// Problem constants (match reference)
constexpr int B_   = 16;
constexpr int C_   = 3;
constexpr int H_   = 224;
constexpr int W_   = 224;
constexpr int OC   = 64;
constexpr int HOUT = 222;
constexpr int WOUT = 222;
constexpr int HW   = HOUT * WOUT;          // 49284
constexpr int DE   = 45;                   // C(9+1,2) pairs per channel
constexpr int KTOT = C_ * DE;              // 135
constexpr int NTILES = (B_ * HW) / 16;     // 49284 (788544 pixels / 16)
constexpr int WAVES_PER_BLOCK = 8;         // 256 threads = 8 wave32

// combinations_with_replacement(range(9), 2) in lexicographic order
constexpr unsigned char CI[DE] = {
  0,0,0,0,0,0,0,0,0,
  1,1,1,1,1,1,1,1,
  2,2,2,2,2,2,2,
  3,3,3,3,3,3,
  4,4,4,4,4,
  5,5,5,5,
  6,6,6,
  7,7,
  8};
constexpr unsigned char CJ[DE] = {
  0,1,2,3,4,5,6,7,8,
  1,2,3,4,5,6,7,8,
  2,3,4,5,6,7,8,
  3,4,5,6,7,8,
  4,5,6,7,8,
  5,6,7,8,
  6,7,8,
  7,8,
  8};

// ---------------------------------------------------------------------------
// Prep: swizzle W (64 x 135 f32) into WMMA A-fragments (f16) in workspace.
// Fragment f = kk*4 + t (kk = K-step 0..4, t = channel tile 0..3).
// A layout used here (16xK=32): lane L<16 holds row m=L with
//   element u -> K = base+u (u=0..15); lanes >=16 hold the same rows
//   with K shifted by +16 (matches B fragment K order).
// Layout in ws: frag[f][lane][u] halves, 16 halves (32 B) contiguous per lane.
// ---------------------------------------------------------------------------
__global__ void prep_weight_frags(const float* __restrict__ Wt,
                                  _Float16* __restrict__ wfrag) {
  int tid = blockIdx.x * blockDim.x + threadIdx.x;
  if (tid >= 20 * 32) return;
  int f    = tid >> 5;
  int lane = tid & 31;
  int kk   = f >> 2;
  int t    = f & 3;
  int hioff = (lane & 16) ? 16 : 0;
  int o = t * 16 + (lane & 15);
  _Float16* dst = wfrag + (size_t)f * 32 * 16 + (size_t)lane * 16;
#pragma unroll
  for (int u = 0; u < 16; ++u) {
    int k = kk * 32 + u + hioff;
    float v = (k < KTOT) ? Wt[o * KTOT + k] : 0.0f;
    dst[u] = (_Float16)v;
  }
}

// ---------------------------------------------------------------------------
// Build this lane's B-fragment slice for K-step KK.
// B layout used here (K=32 x N=16): lane n<16 holds column N=n with K=base+0..15
// (element u -> K=base+u, packed 2/VGPR); lane n+16 holds K=base+16..31.
// All indices are compile-time constants after unrolling, so CI/CJ lookups
// fold to immediates and patch[] stays fully in registers.
// ---------------------------------------------------------------------------
template <int KK, int HIOFF>   // HIOFF = 0 (lanes 0-15) or 16 (lanes 16-31)
__device__ __forceinline__ v16h build_feat(const float (&patch)[3][9]) {
  v16h r;
#pragma unroll
  for (int u = 0; u < 16; ++u) {
    const int k = KK * 32 + u + HIOFF;
    float v = 0.0f;
    if (k < KTOT) {
      const int c = k / DE;
      const int p = k - c * DE;
      v = patch[c][CI[p]] * patch[c][CJ[p]];
    }
    r[u] = (_Float16)v;
  }
  return r;
}

template <int KK>
__device__ __forceinline__ void gemm_step(const float (&patch)[3][9], bool hi,
                                          const v16h* __restrict__ afrags,
                                          int lane, v8f (&acc)[4]) {
  v16h b;
  if (hi) b = build_feat<KK, 16>(patch);
  else    b = build_feat<KK, 0>(patch);
#pragma unroll
  for (int t = 0; t < 4; ++t) {
    v16h a = afrags[(KK * 4 + t) * 32 + lane];
    acc[t] = __builtin_amdgcn_wmma_f32_16x16x32_f16(
        /*neg_a=*/false, a, /*neg_b=*/false, b,
        /*c_mod=*/(short)0, acc[t], /*reuse_a=*/false, /*reuse_b=*/false);
  }
}

// ---------------------------------------------------------------------------
// Main kernel: each wave computes a 16-pixel x 64-channel output tile with
// 20 v_wmma_f32_16x16x32_f16. D layout: lane L -> pixel N = L&15 (coalesced
// stores over l), VGPR j -> channel M = j + (L>=16 ? 8 : 0) within tile.
// ---------------------------------------------------------------------------
__global__ __launch_bounds__(32 * WAVES_PER_BLOCK)
void de_conv_wmma(const float* __restrict__ x,
                  const _Float16* __restrict__ wfrag,
                  float* __restrict__ out) {
  const int lane = threadIdx.x & 31;
  const int wave = threadIdx.x >> 5;
  const int tile = blockIdx.x * WAVES_PER_BLOCK + wave;
  if (tile >= NTILES) return;                 // wave-uniform guard

  const int pix = tile * 16 + (lane & 15);    // this lane's pixel (col of B)
  const int b = pix / HW;
  const int l = pix - b * HW;
  const int h = l / WOUT;
  const int w = l - h * WOUT;
  const bool hi = lane >= 16;

  // Load the 3x3x3 input patch for this pixel (both lane-halves of a pixel
  // load the same patch; L0/L2 absorb the overlap).
  float patch[3][9];
  const float* xb = x + (size_t)b * (C_ * H_ * W_);
#pragma unroll
  for (int c = 0; c < 3; ++c) {
#pragma unroll
    for (int r = 0; r < 3; ++r) {
      const float* p = xb + ((size_t)(c * H_ + h + r)) * W_ + w;
      patch[c][r * 3 + 0] = p[0];
      patch[c][r * 3 + 1] = p[1];
      patch[c][r * 3 + 2] = p[2];
    }
  }

  v8f acc[4] = {};
  const v16h* afr = (const v16h*)wfrag;
  gemm_step<0>(patch, hi, afr, lane, acc);
  gemm_step<1>(patch, hi, afr, lane, acc);
  gemm_step<2>(patch, hi, afr, lane, acc);
  gemm_step<3>(patch, hi, afr, lane, acc);
  gemm_step<4>(patch, hi, afr, lane, acc);

  // Store: out[((b*64 + o) * HW) + l]; lanes 0-15 write consecutive l for a
  // fixed o -> 64B-coalesced half-wave stores (output stream dominates BW).
  const int hoff = hi ? 8 : 0;
  float* ob = out + (size_t)b * OC * HW + l;
#pragma unroll
  for (int t = 0; t < 4; ++t) {
#pragma unroll
    for (int j = 0; j < 8; ++j) {
      ob[(size_t)(t * 16 + j + hoff) * HW] = acc[t][j];
    }
  }
}

extern "C" void kernel_launch(void* const* d_in, const int* in_sizes, int n_in,
                              void* d_out, int out_size, void* d_ws, size_t ws_size,
                              hipStream_t stream) {
  const float* x  = (const float*)d_in[0];   // (16, 3, 224, 224) f32
  const float* Wt = (const float*)d_in[1];   // (64, 135) f32
  float* out = (float*)d_out;                // (16, 64, 222, 222) f32
  _Float16* wfrag = (_Float16*)d_ws;         // 20 frags * 32 lanes * 16 halves = 20 KB

  prep_weight_frags<<<10, 64, 0, stream>>>(Wt, wfrag);

  const int blocks = (NTILES + WAVES_PER_BLOCK - 1) / WAVES_PER_BLOCK;
  de_conv_wmma<<<blocks, 32 * WAVES_PER_BLOCK, 0, stream>>>(x, wfrag, out);
}
